// SpatioTemporalCrossAttention_12902081757864
// MI455X (gfx1250) — matrix operand order repaired and checked by reading, not compiled, not run on hardware
//
#include <hip/hip_runtime.h>
#include <hip/hip_bf16.h>
#include <math.h>

// Problem constants (match reference)
#define Bn 128
#define Tn 16
#define TOn 32
#define Jn 25
#define INn 256
#define OUTn 512
#define Hn 8

typedef __attribute__((ext_vector_type(16))) __bf16 v16bf;
typedef __attribute__((ext_vector_type(8)))  float  v8f;
typedef __attribute__((ext_vector_type(8)))  unsigned short ushort8;

union Frag { ushort8 h[2]; v16bf v; };

__device__ __forceinline__ unsigned short f2bf(float f) {
  unsigned int u = __float_as_uint(f);
  u += 0x7FFFu + ((u >> 16) & 1u);   // round-to-nearest-even
  return (unsigned short)(u >> 16);
}

// ---------------------------------------------------------------------------
// Kernel 1: reduce Wq/Wk over 32-channel groups (channel-sum commutes with the
// projection), convert group weights + full Wv to bf16.
// ---------------------------------------------------------------------------
__global__ void prep_kernel(const float* __restrict__ Wq, const float* __restrict__ bq,
                            const float* __restrict__ Wk, const float* __restrict__ bk,
                            const float* __restrict__ Wv,
                            unsigned short* __restrict__ wqg, float* __restrict__ bqg,
                            unsigned short* __restrict__ wkg, float* __restrict__ bkg,
                            unsigned short* __restrict__ wvb) {
  int tid = blockIdx.x * blockDim.x + threadIdx.x;
  int stride = gridDim.x * blockDim.x;
  for (int i = tid; i < OUTn * INn; i += stride) wvb[i] = f2bf(Wv[i]);
  for (int i = tid; i < 16 * INn; i += stride) {
    int g = i / INn, c = i % INn;
    float sq = 0.f, sk = 0.f;
    for (int o = 0; o < 32; ++o) {
      sq += Wq[(g * 32 + o) * INn + c];
      sk += Wk[(g * 32 + o) * INn + c];
    }
    wqg[i] = f2bf(sq);
    wkg[i] = f2bf(sk);
  }
  for (int i = tid; i < 16; i += stride) {
    float sq = 0.f, sk = 0.f;
    for (int o = 0; o < 32; ++o) { sq += bq[i * 32 + o]; sk += bk[i * 32 + o]; }
    bqg[i] = sq; bkg[i] = sk;
  }
}

// ---------------------------------------------------------------------------
// Kernel 2: convert x / encoder_output activations to bf16.
// ---------------------------------------------------------------------------
__global__ void convert_kernel(const float* __restrict__ x, const float* __restrict__ enc,
                               unsigned short* __restrict__ Xb, unsigned short* __restrict__ Eb) {
  const long NX = (long)Bn * TOn * Jn * INn;   // 26,214,400
  const long NE = (long)Bn * Tn  * Jn * INn;   // 13,107,200
  long tid = (long)blockIdx.x * blockDim.x + threadIdx.x;
  long stride = (long)gridDim.x * blockDim.x;
  for (long i = tid; i < NX + NE; i += stride) {
    if (i < NX) Xb[i] = f2bf(x[i]);
    else        Eb[i - NX] = f2bf(enc[i - NX]);
  }
}

// ---------------------------------------------------------------------------
// WMMA GEMM, N=16:  C(M,16) = A(M,256)_bf16 @ Wg(16,256)_bf16^T + bias(16)
// One wave per 16x16 output tile; fragments built per ISA 16-bit layouts:
//   A lane L: M=L&15; lanes<16 K in {kk+0..7, kk+16..23}, lanes>=16 +8
//   B lane L: N=L&15; same K order as A
// ---------------------------------------------------------------------------
__global__ void gemm_n16_kernel(const unsigned short* __restrict__ Ab,
                                const unsigned short* __restrict__ Wg,
                                const float* __restrict__ bias,
                                float* __restrict__ C, int mtiles) {
  int w = threadIdx.x >> 5;
  int lane = threadIdx.x & 31;
  int tile = blockIdx.x * 8 + w;
  if (tile >= mtiles) return;
  int row0 = tile * 16;
  int col  = lane & 15;
  int hi   = lane >> 4;
  int aoff = hi ? 8 : 0;
  int boff = hi ? 8 : 0;
  const unsigned short* arow = Ab + (long)(row0 + col) * INn;
  const unsigned short* brow = Wg + (long)col * INn;
  v8f acc = {0.f, 0.f, 0.f, 0.f, 0.f, 0.f, 0.f, 0.f};
  for (int kk = 0; kk < INn; kk += 32) {
    Frag fa, fb;
    fa.h[0] = *(const ushort8*)(arow + kk + aoff);
    fa.h[1] = *(const ushort8*)(arow + kk + 16 + aoff);
    fb.h[0] = *(const ushort8*)(brow + kk + boff);
    fb.h[1] = *(const ushort8*)(brow + kk + 16 + boff);
    acc = __builtin_amdgcn_wmma_f32_16x16x32_bf16(false, fa.v, false, fb.v,
                                                  (short)0, acc, false, false);
  }
  float bc = bias[col];
  for (int r = 0; r < 8; ++r) {
    int m = row0 + r + 8 * hi;
    C[(long)m * 16 + col] = acc[r] + bc;
  }
}

// ---------------------------------------------------------------------------
// WMMA GEMM: V(51200,512) = Eb(51200,256) @ Wv(512,256)^T + bv
// blockIdx.x = M tile (0..3199), wave -> N tile (blockIdx.y*8 + wave)
// ---------------------------------------------------------------------------
__global__ void gemm_v_kernel(const unsigned short* __restrict__ Eb,
                              const unsigned short* __restrict__ Wvb,
                              const float* __restrict__ bv,
                              float* __restrict__ V) {
  int w = threadIdx.x >> 5;
  int lane = threadIdx.x & 31;
  int row0 = blockIdx.x * 16;
  int col0 = (blockIdx.y * 8 + w) * 16;
  int col  = lane & 15;
  int hi   = lane >> 4;
  int aoff = hi ? 8 : 0;
  int boff = hi ? 8 : 0;
  const unsigned short* arow = Eb  + (long)(row0 + col) * INn;
  const unsigned short* brow = Wvb + (long)(col0 + col) * INn;
  v8f acc = {0.f, 0.f, 0.f, 0.f, 0.f, 0.f, 0.f, 0.f};
  for (int kk = 0; kk < INn; kk += 32) {
    Frag fa, fb;
    fa.h[0] = *(const ushort8*)(arow + kk + aoff);
    fa.h[1] = *(const ushort8*)(arow + kk + 16 + aoff);
    fb.h[0] = *(const ushort8*)(brow + kk + boff);
    fb.h[1] = *(const ushort8*)(brow + kk + 16 + boff);
    acc = __builtin_amdgcn_wmma_f32_16x16x32_bf16(false, fa.v, false, fb.v,
                                                  (short)0, acc, false, false);
  }
  float bc = bv[col0 + col];
  for (int r = 0; r < 8; ++r) {
    int m = row0 + r + 8 * hi;
    V[(long)m * OUTn + col0 + col] = acc[r] + bc;
  }
}

// ---------------------------------------------------------------------------
// Reshape-aware reductions of V.
// Flat-reshape mapping: per-batch offset h*(T*J*64)+t*(J*64)+j*64+c maps to
// original (t' = 2h + t>>3, n = 25*(t&7)+j, j' = n>>3, o = 64*(n&7)+c).
// Vsj[b,h,t,c] = sum_j v_s_pre   (c in [0,32))
// Vtj[b,h,j,c] = sum_t v_t_pre   (channel 32+c)
// ---------------------------------------------------------------------------
__global__ void reduce_vsj_kernel(const float* __restrict__ V, float* __restrict__ Vsj) {
  int idx = blockIdx.x * blockDim.x + threadIdx.x;
  if (idx >= Bn * Hn * Tn * 32) return;
  int c = idx & 31;
  int t = (idx >> 5) & 15;
  int h = (idx >> 9) & 7;
  int b = idx >> 12;
  int tp = 2 * h + (t >> 3);
  int base = 25 * (t & 7);
  float s = 0.f;
  for (int j = 0; j < Jn; ++j) {
    int n = base + j;
    s += V[((long)(b * Tn + tp) * Jn + (n >> 3)) * OUTn + 64 * (n & 7) + c];
  }
  Vsj[idx] = s;
}

__global__ void reduce_vtj_kernel(const float* __restrict__ V, float* __restrict__ Vtj) {
  int idx = blockIdx.x * blockDim.x + threadIdx.x;
  if (idx >= Bn * Hn * Jn * 32) return;
  int c = idx & 31;
  int j = (idx >> 5) % Jn;
  int hb = idx / (32 * Jn);
  int h = hb & 7;
  int b = hb >> 3;
  float s = 0.f;
  for (int t = 0; t < Tn; ++t) {
    int tp = 2 * h + (t >> 3);
    int n = 25 * (t & 7) + j;
    s += V[((long)(b * Tn + tp) * Jn + (n >> 3)) * OUTn + 64 * (n & 7) + 32 + c];
  }
  Vtj[idx] = s;
}

// ---------------------------------------------------------------------------
// Fused attention + output. One block per (b,h). All small intermediates in LDS.
// ---------------------------------------------------------------------------
__global__ void attn_out_kernel(const float* __restrict__ Xg, const float* __restrict__ Kg,
                                const float* __restrict__ Vsj, const float* __restrict__ Vtj,
                                const int* __restrict__ mask_s, const int* __restrict__ mask_t,
                                const float* __restrict__ Wconv, const float* __restrict__ bconv,
                                float* __restrict__ out) {
  __shared__ float qs[TOn * Jn], qt[TOn * Jn];
  __shared__ float ksg[Tn * Jn], ktg[Tn * Jn];
  __shared__ float vt[Jn * 32];
  __shared__ float kssum[TOn * Jn], vssum[TOn * 32];
  __shared__ float diag[TOn * Jn], rsum[Jn * TOn];
  __shared__ float wc[TOn * Tn], bc[TOn];

  int b = blockIdx.x >> 3;
  int h = blockIdx.x & 7;
  int tid = threadIdx.x;
  const float NEG = -1000000000.f;

  for (int i = tid; i < TOn * Tn; i += 256) wc[i] = Wconv[i];
  for (int i = tid; i < TOn; i += 256) bc[i] = bconv[i];
  // q sums (spatial / temporal halves) via reshape mapping into Xg
  for (int i = tid; i < TOn * Jn; i += 256) {
    int to = i / Jn, j = i % Jn;
    int top = 4 * h + (to >> 3);
    int n = 25 * (to & 7) + j;
    long rowq = (long)(b * TOn + top) * Jn + (n >> 3);
    qs[i] = Xg[rowq * 16 + 2 * (n & 7)];
    qt[i] = Xg[rowq * 16 + 2 * (n & 7) + 1];
  }
  // k sums via reshape mapping into Kg
  for (int i = tid; i < Tn * Jn; i += 256) {
    int t = i / Jn, k = i % Jn;
    int tp = 2 * h + (t >> 3);
    int n = 25 * (t & 7) + k;
    long rowk = (long)(b * Tn + tp) * Jn + (n >> 3);
    ksg[i] = Kg[rowk * 16 + 2 * (n & 7)];
    ktg[i] = Kg[rowk * 16 + 2 * (n & 7) + 1];
  }
  for (int i = tid; i < Jn * 32; i += 256)
    vt[i] = Vtj[(long)(b * Hn + h) * Jn * 32 + i];
  __syncthreads();

  // explode(): temporal mixing of k_s sums and v_s sums (+ bias terms)
  for (int i = tid; i < TOn * Jn; i += 256) {
    int to = i / Jn, k = i % Jn;
    float s = 0.f;
    for (int t = 0; t < Tn; ++t) s += wc[to * Tn + t] * ksg[t * Jn + k];
    kssum[i] = s + 32.f * bc[to];   // sum over 32 channels of bconv broadcast
  }
  for (int i = tid; i < TOn * 32; i += 256) {
    int to = i >> 5, c = i & 31;
    float s = 0.f;
    for (int t = 0; t < Tn; ++t)
      s += wc[to * Tn + t] * Vsj[((long)(b * Hn + h) * Tn + t) * 32 + c];
    vssum[i] = s + 25.f * bc[to];   // sum over J=25 of bconv broadcast
  }
  __syncthreads();

  // spatial softmax: only the diagonal of attn_s survives
  for (int i = tid; i < TOn * Jn; i += 256) {
    int to = i / Jn, j = i % Jn;
    float q = qs[i];
    float m = -3.4e38f;
    for (int k = 0; k < Jn; ++k) {
      float v = mask_s[(to * Jn + j) * Jn + k] ? q * kssum[to * Jn + k] : NEG;
      m = fmaxf(m, v);
    }
    float Z = 0.f, ejj = 0.f;
    for (int k = 0; k < Jn; ++k) {
      float v = mask_s[(to * Jn + j) * Jn + k] ? q * kssum[to * Jn + k] : NEG;
      float e = expf(v - m);
      Z += e;
      if (k == j) ejj = e;
    }
    diag[i] = ejj / Z;
  }
  // temporal softmax row-sums (faithful elementwise divide-then-sum)
  for (int i = tid; i < Jn * TOn; i += 256) {
    int j = i / TOn, to = i % TOn;
    float q = qt[to * Jn + j];
    float m = -3.4e38f;
    for (int t = 0; t < Tn; ++t) {
      float v = mask_t[(j * TOn + to) * Tn + t] ? q * ktg[t * Jn + j] : NEG;
      m = fmaxf(m, v);
    }
    float Z = 0.f;
    for (int t = 0; t < Tn; ++t) {
      float v = mask_t[(j * TOn + to) * Tn + t] ? q * ktg[t * Jn + j] : NEG;
      Z += expf(v - m);
    }
    float invZ = 1.f / Z, s = 0.f;
    for (int t = 0; t < Tn; ++t) {
      float v = mask_t[(j * TOn + to) * Tn + t] ? q * ktg[t * Jn + j] : NEG;
      s += expf(v - m) * invZ;
    }
    rsum[i] = s;
  }
  __syncthreads();

  // output: out[b,to,j, h*64+c]; c<32 spatial, c>=32 temporal
  for (int i = tid; i < TOn * Jn * 64; i += 256) {
    int to = i / (Jn * 64);
    int rem = i % (Jn * 64);
    int j = rem / 64, c = rem % 64;
    float val = (c < 32) ? diag[to * Jn + j] * vssum[to * 32 + c]
                         : rsum[j * TOn + to] * vt[j * 32 + (c - 32)];
    out[((long)(b * TOn + to) * Jn + j) * OUTn + h * 64 + c] = val;
  }
}

// ---------------------------------------------------------------------------
extern "C" void kernel_launch(void* const* d_in, const int* in_sizes, int n_in,
                              void* d_out, int out_size, void* d_ws, size_t ws_size,
                              hipStream_t stream) {
  const float* x     = (const float*)d_in[0];
  const float* enc   = (const float*)d_in[1];
  const int*  mask_s = (const int*)d_in[2];
  const int*  mask_t = (const int*)d_in[3];
  const float* Wq = (const float*)d_in[4];
  const float* bq = (const float*)d_in[5];
  const float* Wk = (const float*)d_in[6];
  const float* bk = (const float*)d_in[7];
  const float* Wv = (const float*)d_in[8];
  const float* bv = (const float*)d_in[9];
  const float* Wconv = (const float*)d_in[10];
  const float* bconv = (const float*)d_in[11];
  float* out = (float*)d_out;

  char* ws = (char*)d_ws;
  size_t off = 0;
  auto carve = [&](size_t bytes) -> char* {
    char* p = ws + off;
    off = (off + bytes + 255) & ~(size_t)255;
    return p;
  };
  unsigned short* Xb  = (unsigned short*)carve(26214400ull * 2);  // bf16 x
  unsigned short* Eb  = (unsigned short*)carve(13107200ull * 2);  // bf16 encoder
  unsigned short* wqg = (unsigned short*)carve(16 * 256 * 2);
  unsigned short* wkg = (unsigned short*)carve(16 * 256 * 2);
  unsigned short* wvb = (unsigned short*)carve(512 * 256 * 2);
  float* bqg = (float*)carve(16 * 4);
  float* bkg = (float*)carve(16 * 4);
  float* Xg  = (float*)carve(102400ull * 16 * 4);
  float* Kg  = (float*)carve(51200ull * 16 * 4);
  float* V   = (float*)carve(51200ull * 512 * 4);
  float* Vsj = (float*)carve(524288ull * 4);
  float* Vtj = (float*)carve(819200ull * 4);

  prep_kernel<<<512, 256, 0, stream>>>(Wq, bq, Wk, bk, Wv, wqg, bqg, wkg, bkg, wvb);
  convert_kernel<<<2048, 256, 0, stream>>>(x, enc, Xb, Eb);
  gemm_n16_kernel<<<800, 256, 0, stream>>>(Xb, wqg, bqg, Xg, 6400);
  gemm_n16_kernel<<<400, 256, 0, stream>>>(Eb, wkg, bkg, Kg, 3200);
  gemm_v_kernel<<<dim3(3200, 4), 256, 0, stream>>>(Eb, wvb, bv, V);
  reduce_vsj_kernel<<<(Bn * Hn * Tn * 32 + 255) / 256, 256, 0, stream>>>(V, Vsj);
  reduce_vtj_kernel<<<(Bn * Hn * Jn * 32 + 255) / 256, 256, 0, stream>>>(V, Vtj);
  attn_out_kernel<<<Bn * Hn, 256, 0, stream>>>(Xg, Kg, Vsj, Vtj, mask_s, mask_t,
                                               Wconv, bconv, out);
}
